// PhasedLSTM_46523085750303
// MI455X (gfx1250) — matrix operand order repaired and checked by reading, not compiled, not run on hardware
//
#include <hip/hip_runtime.h>
#include <hip/hip_bf16.h>

// Phased LSTM for MI455X (gfx1250): persistent grid, LDS-resident bf16 weights,
// v_wmma_f32_16x16x32_bf16 recurrent GEMM, register-resident c/h state.

#define Bsz   256
#define Tsz   1024
#define Hsz   512
#define NWG   64      // 4 batch tiles * 16 hidden slices
#define BT    64      // batch rows per WG
#define HS    32      // hidden units per WG (=> 128 gate columns)
#define LDSK  520     // padded K pitch (bf16 elems) for weight LDS rows

static constexpr float LEAK = 0.001f;
static constexpr float RON  = 0.05f;

typedef __attribute__((ext_vector_type(16))) __bf16 bf16x16;
typedef __attribute__((ext_vector_type(8)))  float  f32x8;

union V16 { bf16x16 v; uint4 q[2]; };

__device__ __forceinline__ unsigned short f2bf(float f) {
    unsigned u = __float_as_uint(f);
    u += 0x7FFFu + ((u >> 16) & 1u);          // round-to-nearest-even
    return (unsigned short)(u >> 16);
}
__device__ __forceinline__ float sigmoid_f(float x) {
    return 1.0f / (1.0f + __expf(-x));
}
__device__ __forceinline__ float tanh_f(float x) {
    float e = __expf(2.0f * x);
    return (e - 1.0f) / (e + 1.0f);
}

__global__ void __launch_bounds__(256, 1)
plstm_persistent(const float* __restrict__ inputs,      // (B,T,4)
                 const int*   __restrict__ seq_starts,  // (B,)
                 const float* __restrict__ w_ih,        // (4H,3)
                 const float* __restrict__ w_hh,        // (4H,H)
                 const float* __restrict__ b_ih,        // (4H,)
                 const float* __restrict__ b_hh,        // (4H,)
                 const float* __restrict__ tau,         // (H,)
                 const float* __restrict__ phase,       // (H,)
                 float* __restrict__ out,               // (B,H)
                 unsigned short* __restrict__ hbuf,     // (B,H) bf16, zero-init
                 unsigned int* __restrict__ bar)        // grid barrier counter
{
    extern __shared__ char smem_raw[];
    unsigned short* wlds = (unsigned short*)smem_raw;                 // [128][LDSK] bf16
    float4*         s_in = (float4*)(smem_raw + 128 * LDSK * 2);      // [BT]

    const int wg    = blockIdx.x;
    const int btile = wg & 3;          // 4 batch tiles
    const int slice = wg >> 2;         // 16 hidden slices
    const int Bb    = btile * BT;
    const int Jb    = slice * HS;

    const int tid   = threadIdx.x;
    const int lane  = tid & 31;
    const int wave  = tid >> 5;
    const int lo    = lane & 15;
    const int hi    = lane >> 4;
    const int mtile = wave & 3;        // 16-row M tile
    const int jhalf = wave >> 2;       // which 16 of the 32 hidden units

    // ---- one-time: convert this WG's w_hh slice to bf16 into LDS ----
    // local column c in 0..127: gate g = c>>5, jj = c&31 ; global row g*H + Jb + jj
    {
        const int c  = tid >> 1;
        const int g  = c >> 5, jj = c & 31;
        const int kh = (tid & 1) * 256;
        const float* src = w_hh + (size_t)(g * Hsz + Jb + jj) * Hsz + kh;
        unsigned short* dst = wlds + c * LDSK + kh;
        #pragma unroll 4
        for (int k = 0; k < 256; k += 4) {
            float4 v = *(const float4*)(src + k);
            dst[k + 0] = f2bf(v.x); dst[k + 1] = f2bf(v.y);
            dst[k + 2] = f2bf(v.z); dst[k + 3] = f2bf(v.w);
        }
    }

    // ---- per-lane constants ----
    const int jg = Jb + jhalf * 16 + lo;            // global hidden index for this lane
    float wi0[4], wi1[4], wi2[4], bsum[4];
    #pragma unroll
    for (int g = 0; g < 4; ++g) {
        const int cg = g * Hsz + jg;
        wi0[g]  = w_ih[cg * 3 + 0];
        wi1[g]  = w_ih[cg * 3 + 1];
        wi2[g]  = w_ih[cg * 3 + 2];
        bsum[g] = b_ih[cg] + b_hh[cg];
    }
    const float tauj = tau[jg];
    const float phj  = phase[jg];
    const float itau = 1.0f / tauj;

    const int brow0 = Bb + mtile * 16;
    int ss[8];
    #pragma unroll
    for (int r = 0; r < 8; ++r) ss[r] = seq_starts[brow0 + 8 * hi + r];

    // register-resident state: element r <-> (batch = brow0 + 8*hi + r, hidden = jg)
    f32x8 creg, hreg;
    #pragma unroll
    for (int r = 0; r < 8; ++r) { creg[r] = 0.0f; hreg[r] = 0.0f; }

    const unsigned short* hrow = hbuf + (size_t)(brow0 + lo) * Hsz;   // A-operand rows
    unsigned int bar_target = 0;

    for (int t = 0; t < Tsz; ++t) {
        // stage this step's input vec4 for the WG's 64 batch rows
        if (tid < BT)
            s_in[tid] = *(const float4*)(inputs + ((size_t)(Bb + tid) * Tsz + t) * 4);
        __syncthreads();

        // C init = x @ w_ih^T + b_ih + b_hh  (K=3, scalar FMAs), cache timestamps
        f32x8 acc[4];
        float tsr[8];
        #pragma unroll
        for (int r = 0; r < 8; ++r) {
            float4 xin = s_in[mtile * 16 + 8 * hi + r];
            tsr[r] = xin.z;                              // channel 2 = timestamp
            #pragma unroll
            for (int g = 0; g < 4; ++g)
                acc[g][r] = fmaf(xin.x, wi0[g],
                            fmaf(xin.y, wi1[g],
                            fmaf(xin.w, wi2[g], bsum[g])));
        }

        // recurrent GEMM: h(16x512 bf16) x w_hh slice, K in chunks of 32
        for (int kb = 0; kb < Hsz; kb += 32) {
            V16 a;
            a.q[0] = *(const uint4*)(hrow + kb + 8 * hi);        // K = 8*hi .. +7
            a.q[1] = *(const uint4*)(hrow + kb + 16 + 8 * hi);   // K = 16+8*hi .. +7
            #pragma unroll
            for (int g = 0; g < 4; ++g) {
                const unsigned short* wr =
                    wlds + (g * 32 + jhalf * 16 + lo) * LDSK + kb;
            V16 b;
            b.q[0] = *(const uint4*)(wr + 8 * hi);
            b.q[1] = *(const uint4*)(wr + 16 + 8 * hi);
                acc[g] = __builtin_amdgcn_wmma_f32_16x16x32_bf16(
                    false, a.v, false, b.v, (short)0, acc[g], false, false);
            }
        }

        // elementwise: gates, mask blend, phased time gate — all in registers
        #pragma unroll
        for (int r = 0; r < 8; ++r) {
            float ig = sigmoid_f(acc[0][r]);
            float fg = sigmoid_f(acc[1][r]);
            float gg = tanh_f(acc[2][r]);
            float og = sigmoid_f(acc[3][r]);
            float c0 = creg[r], h0 = hreg[r];
            float c  = fg * c0 + ig * gg;
            float h  = og * tanh_f(c);
            float m  = (t >= ss[r]) ? 1.0f : 0.0f;
            h = h * m + h0 * (1.0f - m);
            c = c * m + c0 * (1.0f - m);
            float d   = tsr[r] - phj;
            float phi = fabsf(d - tauj * truncf(d * itau)) * itau;   // |fmod(d,tau)|/tau
            float ku  = phi * (2.0f / RON);
            float k   = (phi < RON) ? (2.0f - ku) : (LEAK * phi);
            k = (phi < 0.5f * RON) ? ku : k;
            creg[r] = k * c + (1.0f - k) * c0;
            hreg[r] = k * h + (1.0f - k) * h0;
        }

        // publish h (bf16) for next step's A operand
        #pragma unroll
        for (int r = 0; r < 8; ++r)
            hbuf[(size_t)(brow0 + 8 * hi + r) * Hsz + jg] = f2bf(hreg[r]);

        // grid-wide barrier (monotonic counter; zeroed by memset each launch)
        __threadfence();
        __syncthreads();
        bar_target += NWG;
        if (tid == 0) {
            atomicAdd(bar, 1u);
            while (*(volatile unsigned int*)bar < bar_target)
                __builtin_amdgcn_s_sleep(2);
        }
        __syncthreads();
        __threadfence();
    }

    // final hidden state, fp32
    #pragma unroll
    for (int r = 0; r < 8; ++r)
        out[(size_t)(brow0 + 8 * hi + r) * Hsz + jg] = hreg[r];
}

extern "C" void kernel_launch(void* const* d_in, const int* in_sizes, int n_in,
                              void* d_out, int out_size, void* d_ws, size_t ws_size,
                              hipStream_t stream) {
    const float* inputs  = (const float*)d_in[0];
    const int*   seqs    = (const int*)d_in[1];
    const float* w_ih    = (const float*)d_in[2];
    const float* w_hh    = (const float*)d_in[3];
    const float* b_ih    = (const float*)d_in[4];
    const float* b_hh    = (const float*)d_in[5];
    const float* tau     = (const float*)d_in[6];
    const float* phase   = (const float*)d_in[7];
    float* out = (float*)d_out;

    unsigned int*   bar  = (unsigned int*)d_ws;
    unsigned short* hbuf = (unsigned short*)((char*)d_ws + 256);

    // zero barrier counter + initial h state (bf16 zeros) — capture-safe
    hipMemsetAsync(d_ws, 0, 256 + (size_t)Bsz * Hsz * sizeof(unsigned short), stream);

    const size_t lds_bytes = (size_t)128 * LDSK * 2 + (size_t)BT * sizeof(float4);
    hipFuncSetAttribute((const void*)plstm_persistent,
                        hipFuncAttributeMaxDynamicSharedMemorySize, (int)lds_bytes);

    hipLaunchKernelGGL(plstm_persistent, dim3(NWG), dim3(256), lds_bytes, stream,
                       inputs, seqs, w_ih, w_hh, b_ih, b_hh, tau, phase,
                       out, hbuf, bar);
}
